// S4Decoder_78709570666714
// MI455X (gfx1250) — matrix operand order repaired and checked by reading, not compiled, not run on hardware
//
#include <hip/hip_runtime.h>
#include <hip/hip_bf16.h>
#include <math.h>
#include <stdint.h>

// ---------------------------------------------------------------------------
// S4D decoder for MI455X (gfx1250, wave32, WMMA + Tensor Data Mover).
//  B=32, D=H=256, L=1024, N=64, NLAYERS=4
//  - FFT conv replaced by the mathematically identical diagonal SSM recurrence
//  - all GEMMs via v_wmma_f32_16x16x32_bf16 (bf16 inputs, f32 accumulate)
//  - GEMM tiles staged into LDS by TDM (tensor_load_to_lds), double-buffered,
//    synchronized with s_wait_tensorcnt; all B operands stored K-contiguous
//    so no software transpose exists anywhere in the hot loops.
//  - ~100 MB workspace: resident in the 192MB L2.
// ---------------------------------------------------------------------------

#define Bc 32
#define Dc 256
#define Lc 1024
#define Nc 64
#define NL 4
#define TPITCH 40   // staged LDS row pitch in bf16 elems: 32 data + 4-DWORD pad

typedef __attribute__((ext_vector_type(16))) __bf16 v16bf;
typedef __attribute__((ext_vector_type(8)))  float  v8f;
typedef __attribute__((ext_vector_type(4)))  unsigned int u32x4;
typedef __attribute__((ext_vector_type(8)))  int i32x8;
typedef __attribute__((ext_vector_type(4)))  int i32x4;

union FragBF { v16bf v; uint4 q[2]; __bf16 e[16]; };
union AccF   { v8f   v; float e[8]; };

#if __has_builtin(__builtin_amdgcn_tensor_load_to_lds)
#define USE_TDM 1
#else
#define USE_TDM 0
#endif

#if USE_TDM
__device__ __forceinline__ unsigned ldsOff(const void* p) {
    // generic LDS address: low 32 bits are the LDS byte offset
    return (unsigned)(size_t)p;
}
// 2D TDM load: 64 rows x 32 bf16, row stride `pitch` elems in global,
// LDS pitch = 32 elems + 4 DWORDs pad (= TPITCH) via D# pad fields.
__device__ __forceinline__ void tdm_load_tile64x32(const __bf16* gsrc,
                                                   unsigned lds_byte_off,
                                                   int pitch_elems) {
    unsigned long long ga = (unsigned long long)(size_t)gsrc;
    u32x4 g0;
    g0[0] = 1u;                                   // count=1 valid descriptor
    g0[1] = lds_byte_off;                         // lds_addr
    g0[2] = (unsigned)ga;                         // global_addr[31:0]
    g0[3] = (unsigned)((ga >> 32) & 0x1ffffffull) // global_addr[56:32]
          | (2u << 30);                           // type=2 ("image")
    i32x8 g1;
    g1[0] = (int)((1u << 16)        // data_size = 1 -> 2 bytes (bf16)
                | (1u << 20)        // pad_enable
                | (3u << 22)        // pad_interval: 16 DWORDs of data
                | (3u << 25));      // pad_amount:   4 DWORDs of pad
    const unsigned td0 = 0x10000u, td1 = 0x10000u;   // tensor dims (no clip)
    g1[1] = (int)((td0 & 0xffffu) << 16);            // barrier_addr=0 | td0.lo
    g1[2] = (int)(((td0 >> 16) & 0xffffu) | ((td1 & 0xffffu) << 16));
    g1[3] = (int)(((td1 >> 16) & 0xffffu) | (32u << 16));  // tile_dim0 = 32
    g1[4] = 64;                                          // tile_dim1=64, dim2=0
    g1[5] = pitch_elems;                                 // tensor_dim0_stride lo
    g1[6] = 0;
    g1[7] = 0;
    i32x4 z4 = (i32x4){0, 0, 0, 0};
    i32x8 z8 = (i32x8){0, 0, 0, 0, 0, 0, 0, 0};
    // clang-23 / therock-10.0 6-arg form: (g0, g1, g2, g3, g4, cpol)
    __builtin_amdgcn_tensor_load_to_lds(g0, g1, z4, z4, z8, 0);
}
#endif

// ------------------------------ prep kernels -------------------------------

// lambda = exp(dt*A), Ctilde = 2*C*(exp(dt*A)-1)/A   (complex, per layer,h,n)
__global__ void s4d_prep_ssm(const float* __restrict__ log_dt,
                             const float* __restrict__ logA_real,
                             const float* __restrict__ A_imag,
                             const float* __restrict__ C_re,
                             const float* __restrict__ C_im,
                             float* __restrict__ lam, float* __restrict__ ct) {
    int idx = blockIdx.x * 256 + threadIdx.x;
    if (idx >= NL * Dc * Nc) return;
    int ih = idx / Nc;
    int h  = ih % Dc, i = ih / Dc;
    float dt = __expf(log_dt[i * Dc + h]);
    float Ar = -__expf(logA_real[idx]);
    float Ai = A_imag[idx];
    float er = __expf(Ar * dt);
    float lr = er * __cosf(Ai * dt), li = er * __sinf(Ai * dt);
    float dr = lr - 1.f, di = li;                 // exp(dtA) - 1
    float den = Ar * Ar + Ai * Ai;
    float wr = (dr * Ar + di * Ai) / den;         // (exp(dtA)-1)/A
    float wi = (di * Ar - dr * Ai) / den;
    float cr = C_re[idx], ci = C_im[idx];
    lam[2 * idx] = lr;  lam[2 * idx + 1] = li;
    ct[2 * idx]  = 2.f * (cr * wr - ci * wi);
    ct[2 * idx + 1] = 2.f * (cr * wi + ci * wr);
}

__global__ void s4d_cvt_bf16(const float* __restrict__ src,
                             __bf16* __restrict__ dst, int n) {
    int idx = blockIdx.x * 256 + threadIdx.x;
    if (idx < n) dst[idx] = (__bf16)src[idx];
}

// dst[c*R + r] = src[r*C + c]
__global__ void s4d_cvt_transpose(const float* __restrict__ src,
                                  __bf16* __restrict__ dst, int R, int C) {
    int idx = blockIdx.x * 256 + threadIdx.x;
    if (idx >= R * C) return;
    int r = idx / C, c = idx % C;
    dst[c * R + r] = (__bf16)src[idx];
}

// Xin[b][l][d] = bf16(embed[b,d] + pos_enc(l,d))   (layout (B,L,D): K-contig)
__global__ void s4d_build_xin(const float* __restrict__ embed,
                              __bf16* __restrict__ xin) {
    int idx = blockIdx.x * 256 + threadIdx.x;   // (b*L + l)*D + d
    int d = idx & (Dc - 1);
    int bl = idx >> 8;
    int l = bl & (Lc - 1), b = bl >> 10;
    const float k = -logf(10000.f) / (float)Dc;
    float div = __expf((float)(d & ~1) * k);
    float ph = (float)l * div;
    float pe = (d & 1) ? __cosf(ph) : __sinf(ph);
    xin[idx] = (__bf16)(embed[b * Dc + d] + pe);
}

// ------------------------------ WMMA GEMM ----------------------------------
// C(M,N) = A(M,K) * Bt(N,K)^T (+bias); A,Bt bf16 row-major (both K-contig).
// GLU==1: also computes gate rows A[m+256,:], epilogue a*sigmoid(g).
// Block 128 thr (4 waves), block tile 64x64, wave tile 32x32, K-tile 32.
// TDM path: wave0 DMAs the B tile, wave1 the A tile, wave2 the gate tile;
// double-buffered stages, s_wait_tensorcnt + workgroup barrier.
template <int GLU, int ROWBIAS>
__global__ __launch_bounds__(128) void s4d_gemm_wmma(
    const __bf16* __restrict__ A, int lda, long aBatch,
    const __bf16* __restrict__ Bt, int ldb, long bBatch,
    float* __restrict__ C, int ldc, long cBatch,
    const float* __restrict__ bias, int K) {
    __shared__ __align__(16) __bf16 sA[2][2][64][TPITCH];  // [stage][glu half]
    __shared__ __align__(16) __bf16 sB[2][64][TPITCH];     // [stage][n][k]

    const int tid  = threadIdx.x;
    const int lane = tid & 31, wave = tid >> 5;
    const int wm = wave >> 1, wn = wave & 1;
    const long bz = blockIdx.z;
    const __bf16* Ab = A + bz * aBatch;
    const __bf16* Bb = Bt + bz * bBatch;
    float* Cb = C + bz * cBatch;
    const int m0 = blockIdx.y * 64;
    const int n0 = blockIdx.x * 64;

    AccF acc[2][2], accg[2][2];
#pragma unroll
    for (int mi = 0; mi < 2; ++mi)
#pragma unroll
        for (int ni = 0; ni < 2; ++ni)
#pragma unroll
            for (int r = 0; r < 8; ++r) { acc[mi][ni].e[r] = 0.f; accg[mi][ni].e[r] = 0.f; }

    const int nk = K / 32;
    const __bf16* aP = Ab + (long)m0 * lda;
    const __bf16* gP = Ab + (long)(m0 + Dc) * lda;
    const __bf16* bP = Bb + (long)n0 * ldb;

#if USE_TDM
    if (wave == 0)              tdm_load_tile64x32(bP, ldsOff(&sB[0][0][0]), ldb);
    else if (wave == 1)         tdm_load_tile64x32(aP, ldsOff(&sA[0][0][0][0]), lda);
    else if (GLU && wave == 2)  tdm_load_tile64x32(gP, ldsOff(&sA[0][1][0][0]), lda);
#endif

    for (int kt = 0; kt < nk; ++kt) {
        const int st = kt & 1;
#if USE_TDM
        const int kn = (kt + 1) * 32;
        if (wave == 0) {
            if (kt + 1 < nk) { tdm_load_tile64x32(bP + kn, ldsOff(&sB[st ^ 1][0][0]), ldb);
                               __builtin_amdgcn_s_wait_tensorcnt(1); }
            else             { __builtin_amdgcn_s_wait_tensorcnt(0); }
        } else if (wave == 1) {
            if (kt + 1 < nk) { tdm_load_tile64x32(aP + kn, ldsOff(&sA[st ^ 1][0][0][0]), lda);
                               __builtin_amdgcn_s_wait_tensorcnt(1); }
            else             { __builtin_amdgcn_s_wait_tensorcnt(0); }
        } else if (GLU && wave == 2) {
            if (kt + 1 < nk) { tdm_load_tile64x32(gP + kn, ldsOff(&sA[st ^ 1][1][0][0]), lda);
                               __builtin_amdgcn_s_wait_tensorcnt(1); }
            else             { __builtin_amdgcn_s_wait_tensorcnt(0); }
        }
        __syncthreads();           // publish stage st to all waves
#else
        // fallback: vector loads + LDS stores into stage 0 (single buffer)
        const int k0 = kt * 32;
        const int row = tid >> 1;             // 0..63
        const int ks  = (tid & 1) * 16;       // 0/16
        const uint4* ap = (const uint4*)(aP + (long)row * lda + k0 + ks);
        uint4 a0 = ap[0], a1 = ap[1];
        const uint4* bp = (const uint4*)(bP + (long)row * ldb + k0 + ks);
        uint4 b0 = bp[0], b1 = bp[1];
        uint4 g0q, g1q;
        if (GLU) { const uint4* gp = (const uint4*)(gP + (long)row * lda + k0 + ks);
                   g0q = gp[0]; g1q = gp[1]; }
        if (k0 + 32 < K) {
            __builtin_prefetch(aP + (long)row * lda + k0 + 32 + ks, 0, 0);
            __builtin_prefetch(bP + (long)row * ldb + k0 + 32 + ks, 0, 0);
        }
        __syncthreads();
        *(uint4*)&sA[0][0][row][ks] = a0; *(uint4*)&sA[0][0][row][ks + 8] = a1;
        *(uint4*)&sB[0][row][ks] = b0;    *(uint4*)&sB[0][row][ks + 8] = b1;
        if (GLU) { *(uint4*)&sA[0][1][row][ks] = g0q; *(uint4*)&sA[0][1][row][ks + 8] = g1q; }
        __syncthreads();
        const int st_ = 0; (void)st;
#define STAGE st_
#endif
#if USE_TDM
#define STAGE st
#endif
        const int ml  = lane & 15;
        const int khA = (lane >> 4) * 8;   // A lanes 0-15: K{0..7,16..23}; 16-31: K{8..15,24..31}
        const int khB = (lane >> 4) * 8;   // B lanes 0-15: K{0..7,16..23}; 16-31: K{8..15,24..31}
        FragBF bf[2];
#pragma unroll
        for (int ni = 0; ni < 2; ++ni) {
            int n = wn * 32 + ni * 16 + ml;
            bf[ni].q[0] = *(const uint4*)&sB[STAGE][n][khB];
            bf[ni].q[1] = *(const uint4*)&sB[STAGE][n][khB + 16];
        }
#pragma unroll
        for (int mi = 0; mi < 2; ++mi) {
            int m = wm * 32 + mi * 16 + ml;
            FragBF af;
            af.q[0] = *(const uint4*)&sA[STAGE][0][m][khA];
            af.q[1] = *(const uint4*)&sA[STAGE][0][m][khA + 16];
#pragma unroll
            for (int ni = 0; ni < 2; ++ni)
                acc[mi][ni].v = __builtin_amdgcn_wmma_f32_16x16x32_bf16(
                    false, af.v, false, bf[ni].v, (short)0, acc[mi][ni].v, false, false);
            if (GLU) {
                FragBF gf;
                gf.q[0] = *(const uint4*)&sA[STAGE][1][m][khA];
                gf.q[1] = *(const uint4*)&sA[STAGE][1][m][khA + 16];
#pragma unroll
                for (int ni = 0; ni < 2; ++ni)
                    accg[mi][ni].v = __builtin_amdgcn_wmma_f32_16x16x32_bf16(
                        false, gf.v, false, bf[ni].v, (short)0, accg[mi][ni].v, false, false);
            }
        }
#undef STAGE
        __syncthreads();           // readers done before stage is overwritten
    }

    // epilogue: lane -> n = lane&15, rows = r + (lane>=16 ? 8 : 0)
    const int ml = lane & 15;
    const int ms = (lane >> 4) * 8;
#pragma unroll
    for (int mi = 0; mi < 2; ++mi)
#pragma unroll
        for (int ni = 0; ni < 2; ++ni) {
            int gm = m0 + wm * 32 + mi * 16 + ms;
            int gn = n0 + wn * 32 + ni * 16 + ml;
#pragma unroll
            for (int r = 0; r < 8; ++r) {
                float v = acc[mi][ni].e[r];
                if (GLU) {
                    float a = v + bias[gm + r];
                    float g = accg[mi][ni].e[r] + bias[gm + r + Dc];
                    v = a * (1.f / (1.f + __expf(-g)));
                } else {
                    v += ROWBIAS ? bias[gm + r] : bias[gn];
                }
                Cb[(long)(gm + r) * ldc + gn] = v;
            }
        }
}

// ------------------------------ SSM scan -----------------------------------
// One wave per (b,h); lane owns 2 complex states (N=64 = 32 lanes * 2).
// Block = 16 waves covering 16 consecutive h of one b, so the GELU output
// can be emitted K-contiguous as (B,L,H) through a small LDS transpose.
__global__ __launch_bounds__(512) void s4d_ssm_scan(
    const float* __restrict__ x, const float* __restrict__ lam,
    const float* __restrict__ ct, const float* __restrict__ dskip,
    __bf16* __restrict__ ylt) {
    __shared__ float stg[16][33];
    const int tid = threadIdx.x;
    const int lane = tid & 31, wave = tid >> 5;
    const int b = blockIdx.x >> 4;
    const int h0 = (blockIdx.x & 15) * 16;
    const int h = h0 + wave;
    const float* xp = x + ((long)b * Dc + h) * Lc;

    const int pidx = (h * Nc + 2 * lane) * 2;
    const float l0r = lam[pidx], l0i = lam[pidx + 1];
    const float l1r = lam[pidx + 2], l1i = lam[pidx + 3];
    const float c0r = ct[pidx], c0i = ct[pidx + 1];
    const float c1r = ct[pidx + 2], c1i = ct[pidx + 3];
    const float ds = dskip[h];

    float s0r = 0.f, s0i = 0.f, s1r = 0.f, s1i = 0.f;
    for (int t0 = 0; t0 < Lc; t0 += 32) {
        float u = xp[t0 + lane];
        float yv = 0.f;
#pragma unroll
        for (int t = 0; t < 32; ++t) {
            float ut = __shfl(u, t, 32);
            float nr = fmaf(s0r, l0r, ut); nr = fmaf(s0i, -l0i, nr);
            float ni = s0r * l0i;          ni = fmaf(s0i, l0r, ni);
            s0r = nr; s0i = ni;
            nr = fmaf(s1r, l1r, ut); nr = fmaf(s1i, -l1i, nr);
            ni = s1r * l1i;          ni = fmaf(s1i, l1r, ni);
            s1r = nr; s1i = ni;
            float p = s0r * c0r;  p = fmaf(s0i, -c0i, p);
            p = fmaf(s1r, c1r, p); p = fmaf(s1i, -c1i, p);
#pragma unroll
            for (int off = 16; off > 0; off >>= 1) p += __shfl_xor(p, off, 32);
            if (lane == t) yv = p;
        }
        float val = yv + u * ds;
        val = 0.5f * val * (1.f + erff(val * 0.70710678118f));   // exact gelu
        stg[wave][lane] = val;
        __syncthreads();
        {   // transpose 16h x 32l -> (B,L,H) bf16, h-contiguous stores
            int hh = tid & 15, ll = tid >> 4;
            ylt[((long)b * Lc + t0 + ll) * Dc + h0 + hh] = (__bf16)stg[hh][ll];
        }
        __syncthreads();
    }
}

// ------------------------ residual + LayerNorm -----------------------------
// v = glu_out + x ; LN over channels (h) at each (b,l); writes x fp32 (B,H,L)
// and, on the last layer, the bf16 transpose (B,L,H) for the decoder GEMM.
__global__ __launch_bounds__(256) void s4d_resid_ln(
    const float* __restrict__ y2, float* __restrict__ x,
    const float* __restrict__ gamma, const float* __restrict__ beta,
    __bf16* __restrict__ xlnt, int writeT) {
    __shared__ float vt[Dc][33];
    __shared__ float red[2][8][32];
    __shared__ float mv[2][32];
    const int b = blockIdx.y;
    const int l0 = blockIdx.x * 32;
    const int tid = threadIdx.x;
    const long base = (long)b * Dc * Lc + l0;

#pragma unroll
    for (int j = 0; j < 8; ++j) {
        int idx = tid + j * 256;             // 2048 float4 loads
        int hh = idx >> 3, lq = (idx & 7) * 4;
        float4 a = *(const float4*)&y2[base + (long)hh * Lc + lq];
        float4 c = *(const float4*)&x[base + (long)hh * Lc + lq];
        vt[hh][lq + 0] = a.x + c.x;  vt[hh][lq + 1] = a.y + c.y;
        vt[hh][lq + 2] = a.z + c.z;  vt[hh][lq + 3] = a.w + c.w;
    }
    __syncthreads();

    const int l = tid & 31, hg = tid >> 5;
    float s = 0.f, s2 = 0.f;
    for (int h = hg * 32; h < hg * 32 + 32; ++h) {
        float v = vt[h][l]; s += v; s2 += v * v;
    }
    red[0][hg][l] = s; red[1][hg][l] = s2;
    __syncthreads();
    if (tid < 32) {
        float ts = 0.f, ts2 = 0.f;
        for (int g = 0; g < 8; ++g) { ts += red[0][g][tid]; ts2 += red[1][g][tid]; }
        float mu = ts * (1.f / Dc);
        float var = ts2 * (1.f / Dc) - mu * mu;
        mv[0][tid] = mu; mv[1][tid] = rsqrtf(var + 1e-5f);
    }
    __syncthreads();
    const float mu = mv[0][l], rs = mv[1][l];
    for (int h = hg * 32; h < hg * 32 + 32; ++h) {
        float v = (vt[h][l] - mu) * rs * gamma[h] + beta[h];
        x[base + (long)h * Lc + l] = v;
        vt[h][l] = v;
    }
    if (writeT) {
        __syncthreads();
#pragma unroll 4
        for (int j = 0; j < 32; ++j)   // thread tid = channel h; coalesced in h
            xlnt[((long)b * Lc + l0 + j) * Dc + tid] = (__bf16)vt[tid][j];
    }
}

// ------------------------------ launcher -----------------------------------
extern "C" void kernel_launch(void* const* d_in, const int* in_sizes, int n_in,
                              void* d_out, int out_size, void* d_ws, size_t ws_size,
                              hipStream_t stream) {
    (void)in_sizes; (void)n_in; (void)out_size; (void)ws_size;
    // setup_inputs order:
    const float* embed     = (const float*)d_in[1];
    const float* enc_w     = (const float*)d_in[2];
    const float* enc_b     = (const float*)d_in[3];
    const float* dec_w     = (const float*)d_in[4];
    const float* dec_b     = (const float*)d_in[5];
    const float* log_dt    = (const float*)d_in[6];
    const float* logA_real = (const float*)d_in[7];
    const float* A_imag    = (const float*)d_in[8];
    const float* C_re      = (const float*)d_in[9];
    const float* C_im      = (const float*)d_in[10];
    const float* D_skip    = (const float*)d_in[11];
    const float* out_w     = (const float*)d_in[12];
    const float* out_b     = (const float*)d_in[13];
    const float* ln_g      = (const float*)d_in[14];
    const float* ln_b      = (const float*)d_in[15];
    float* out = (float*)d_out;

    // workspace layout (256B aligned)
    char* ws = (char*)d_ws;
    size_t off = 0;
    auto take = [&](size_t bytes) { char* p = ws + off; off = (off + bytes + 255) & ~(size_t)255; return p; };
    float* lam    = (float*)take((size_t)NL * Dc * Nc * 2 * 4);
    float* ct     = (float*)take((size_t)NL * Dc * Nc * 2 * 4);
    __bf16* encWT = (__bf16*)take((size_t)Dc * Dc * 2);    // (h,d) row-major
    __bf16* decWT = (__bf16*)take((size_t)Dc * Dc * 2);    // (d,h) row-major
    __bf16* outWb = (__bf16*)take((size_t)NL * 2 * Dc * Dc * 2);
    char*  shared32 = take((size_t)Bc * Dc * Lc * 4);  // xin bf16 16MB, then y2 f32 32MB
    __bf16* xin  = (__bf16*)shared32;                  // (B,L,D)
    float*  y2   = (float*)shared32;                   // (B,H,L) glu output
    float*  xf   = (float*)take((size_t)Bc * Dc * Lc * 4);   // x state (B,H,L) fp32
    __bf16* ylt  = (__bf16*)take((size_t)Bc * Dc * Lc * 2);  // gelu out (B,L,H) bf16
    __bf16* xlnt = (__bf16*)take((size_t)Bc * Lc * Dc * 2);  // final LN (B,L,H) bf16

    // prep
    s4d_prep_ssm<<<(NL * Dc * Nc + 255) / 256, 256, 0, stream>>>(
        log_dt, logA_real, A_imag, C_re, C_im, lam, ct);
    s4d_cvt_transpose<<<(Dc * Dc + 255) / 256, 256, 0, stream>>>(enc_w, encWT, Dc, Dc);
    s4d_cvt_transpose<<<(Dc * Dc + 255) / 256, 256, 0, stream>>>(dec_w, decWT, Dc, Dc);
    s4d_cvt_bf16<<<(NL * 2 * Dc * Dc + 255) / 256, 256, 0, stream>>>(out_w, outWb, NL * 2 * Dc * Dc);
    s4d_build_xin<<<(Bc * Dc * Lc) / 256, 256, 0, stream>>>(embed, xin);

    // encoder: xf[b,h,l] = enc_wT(h,:) . xin[b,l,:] + enc_b[h]
    s4d_gemm_wmma<0, 1><<<dim3(Lc / 64, Dc / 64, Bc), 128, 0, stream>>>(
        encWT, Dc, 0L, xin, Dc, (long)Lc * Dc, xf, Lc, (long)Dc * Lc, enc_b, Dc);

    for (int i = 0; i < NL; ++i) {
        s4d_ssm_scan<<<(Bc * Dc) / 16, 512, 0, stream>>>(
            xf, lam + (long)i * Dc * Nc * 2, ct + (long)i * Dc * Nc * 2,
            D_skip + i * Dc, ylt);
        // z = out_w @ y (+bias), fused GLU -> y2 (B,H,L) fp32
        s4d_gemm_wmma<1, 1><<<dim3(Lc / 64, Dc / 64, Bc), 128, 0, stream>>>(
            outWb + (long)i * 2 * Dc * Dc, Dc, 0L, ylt, Dc, (long)Lc * Dc,
            y2, Lc, (long)Dc * Lc, out_b + i * 2 * Dc, Dc);
        s4d_resid_ln<<<dim3(Lc / 32, Bc), 256, 0, stream>>>(
            y2, xf, ln_g + i * Dc, ln_b + i * Dc, xlnt, (i == NL - 1) ? 1 : 0);
    }

    // decoder: out(BL,D) = xlnt(BL,H) . decWT(D,H)^T + dec_b (per-column bias)
    s4d_gemm_wmma<0, 0><<<dim3(Dc / 64, (Bc * Lc) / 64, 1), 128, 0, stream>>>(
        xlnt, Dc, 0L, decWT, Dc, 0L, out, Dc, 0L, dec_b, Dc);
}
